// Gemma4Attention_70248485093994
// MI455X (gfx1250) — matrix operand order repaired and checked by reading, not compiled, not run on hardware
//
#include <hip/hip_runtime.h>
#include <hip/hip_bf16.h>
#include <stdint.h>

#define SEQ 2048
#define HIDDEN 2048
#define NUM_HEADS 16
#define NUM_KV_HEADS 8
#define HEAD_DIM 256
#define ROT_DIM 64
#define Q_SIZE (NUM_HEADS * HEAD_DIM)     /* 4096 */
#define KV_SIZE (NUM_KV_HEADS * HEAD_DIM) /* 2048 */
#define QKV_COLS (Q_SIZE + 2 * KV_SIZE)   /* 8192 */

typedef __attribute__((ext_vector_type(16))) __bf16 v16bf;
typedef __attribute__((ext_vector_type(8)))  float  v8f;
typedef __attribute__((ext_vector_type(4)))  unsigned int u32x4;
typedef int v4i __attribute__((vector_size(16)));  // matches async builtin param type

union Frag {
  v16bf bf;
  u32x4 q[2];
  unsigned int u[8];
  unsigned short h[16];
};

__device__ __forceinline__ unsigned short f2bf(float f) {
  unsigned int u = __float_as_uint(f);
  u += 0x7FFFu + ((u >> 16) & 1u);  // round-to-nearest-even
  return (unsigned short)(u >> 16);
}

// ---- CDNA5 async global->LDS copy (ASYNCcnt path), with safe fallback ----
__device__ __forceinline__ void async_cp16(const void* g, void* l) {
#if defined(__gfx1250__) && __has_builtin(__builtin_amdgcn_global_load_async_to_lds_b128)
  __builtin_amdgcn_global_load_async_to_lds_b128(
      (__attribute__((address_space(1))) v4i*)(uintptr_t)g,
      (__attribute__((address_space(3))) v4i*)(uintptr_t)l, 0, 0);
#else
  *(u32x4*)l = *(const u32x4*)g;
#endif
}
template <int N>
__device__ __forceinline__ void wait_async() {
#if defined(__gfx1250__)
#if __has_builtin(__builtin_amdgcn_s_wait_asynccnt)
  __builtin_amdgcn_s_wait_asynccnt(N);
#else
  asm volatile("s_wait_asynccnt %0" ::"i"(N) : "memory");
#endif
#endif
}

// A-fragment (16-bit, 16x32): lanes 0-15 K=0-7/16-23, lanes 16-31 K=8-15/24-31
__device__ __forceinline__ Frag lds_frag_a(const unsigned char* rowPtr, int hi) {
  Frag f;
  f.q[0] = *(const u32x4*)(rowPtr + hi * 16);
  f.q[1] = *(const u32x4*)(rowPtr + hi * 16 + 32);
  return f;
}
// B-fragment (16-bit, 32x16): lanes 0-15 K=0-7/16-23, lanes 16-31 K=8-15/24-31
__device__ __forceinline__ Frag lds_frag_b(const unsigned char* rowPtr, int hi) {
  Frag f;
  f.q[0] = *(const u32x4*)(rowPtr + hi * 16);
  f.q[1] = *(const u32x4*)(rowPtr + hi * 16 + 32);
  return f;
}
__device__ __forceinline__ Frag glb_frag_a(const unsigned short* p, int hi) {
  Frag f;
  f.q[0] = *(const u32x4*)(p + hi * 8);
  f.q[1] = *(const u32x4*)(p + hi * 8 + 16);
  return f;
}

// ---------------------------------------------------------------- converts
__global__ __launch_bounds__(256)
void cvt_f32_bf16(const float* __restrict__ in, unsigned short* __restrict__ out, long n) {
  long i0 = ((long)blockIdx.x * 256 + threadIdx.x) * 8;
  if (i0 + 8 > n) return;
  u32x4 a = *(const u32x4*)(in + i0);
  u32x4 b = *(const u32x4*)(in + i0 + 4);
  u32x4 o;
  float* af = (float*)&a;
  float* bf = (float*)&b;
  o[0] = (unsigned int)f2bf(af[0]) | ((unsigned int)f2bf(af[1]) << 16);
  o[1] = (unsigned int)f2bf(af[2]) | ((unsigned int)f2bf(af[3]) << 16);
  o[2] = (unsigned int)f2bf(bf[0]) | ((unsigned int)f2bf(bf[1]) << 16);
  o[3] = (unsigned int)f2bf(bf[2]) | ((unsigned int)f2bf(bf[3]) << 16);
  *(u32x4*)(out + i0) = o;
}

// ------------------------------------------------- GEMM: C[M,N] = A(MxK) * B(NxK)^T
// Double-buffered LDS tiles; async fetch of tile t+1 overlaps WMMA on tile t.
__global__ __launch_bounds__(256)
void gemm_bf16_nt(const unsigned short* __restrict__ A,
                  const unsigned short* __restrict__ B,
                  float* __restrict__ C, int N, int K) {
  const int tid = threadIdx.x;
  const int lane = tid & 31, wave = tid >> 5;
  const int lsel = lane & 15, hi = lane >> 4;
  const int waveM = wave >> 1, waveN = wave & 1;
  const int m0 = blockIdx.y * 128;
  const int n0 = blockIdx.x * 128;

  __shared__ __align__(16) unsigned char As[2][128 * 80];
  __shared__ __align__(16) unsigned char Bs[2][128 * 80];

  // staging: 128 rows x 64B = 512 16B-chunks per tile; 2 chunks/thread/tile
  const int r0 = tid >> 2, c0 = tid & 3;
  const unsigned short* gA0 = A + (size_t)(m0 + r0) * K + c0 * 8;
  const unsigned short* gB0 = B + (size_t)(n0 + r0) * K + c0 * 8;
  const size_t rstep = (size_t)64 * K;  // 64 rows down for second chunk
  const int loff = r0 * 80 + c0 * 16;

  auto stage = [&](int buf, int k0) {
    async_cp16(gA0 + k0, As[buf] + loff);
    async_cp16(gA0 + k0 + rstep, As[buf] + loff + 64 * 80);
    async_cp16(gB0 + k0, Bs[buf] + loff);
    async_cp16(gB0 + k0 + rstep, Bs[buf] + loff + 64 * 80);
  };

  v8f acc[2][4];
#pragma unroll
  for (int a = 0; a < 2; a++)
#pragma unroll
    for (int b = 0; b < 4; b++) acc[a][b] = {};

  stage(0, 0);
  int p = 0;
  for (int k0 = 0; k0 < K; k0 += 32, p ^= 1) {
    if (k0 + 32 < K) {
      stage(p ^ 1, k0 + 32);  // prefetch next tile into the other buffer
      wait_async<4>();        // in-order completion: first 4 (current tile) done
    } else {
      wait_async<0>();
    }
    __syncthreads();

    Frag af[2], bfm[4];
#pragma unroll
    for (int mi = 0; mi < 2; mi++)
      af[mi] = lds_frag_a(As[p] + (waveM * 32 + mi * 16 + lsel) * 80, hi);
#pragma unroll
    for (int ni = 0; ni < 4; ni++)
      bfm[ni] = lds_frag_b(Bs[p] + (waveN * 64 + ni * 16 + lsel) * 80, hi);
#pragma unroll
    for (int mi = 0; mi < 2; mi++)
#pragma unroll
      for (int ni = 0; ni < 4; ni++)
        acc[mi][ni] = __builtin_amdgcn_wmma_f32_16x16x32_bf16(
            false, af[mi].bf, false, bfm[ni].bf, (short)0, acc[mi][ni], false, false);
    __syncthreads();
  }

#pragma unroll
  for (int mi = 0; mi < 2; mi++)
#pragma unroll
    for (int ni = 0; ni < 4; ni++) {
      int col = n0 + waveN * 64 + ni * 16 + lsel;
#pragma unroll
      for (int j = 0; j < 8; j++) {
        int row = m0 + waveM * 32 + mi * 16 + j + 8 * hi;
        C[(size_t)row * N + col] = acc[mi][ni][j];
      }
    }
}

// ---------------------------------------- RMSNorm + RoPE + bf16 pack of Q/K/V
__global__ __launch_bounds__(256)
void norm_rope_kernel(const float* __restrict__ qkv,
                      const float* __restrict__ q_norm_w,
                      const float* __restrict__ k_norm_w,
                      const int* __restrict__ positions,
                      unsigned short* __restrict__ Qb,
                      unsigned short* __restrict__ Kb,
                      unsigned short* __restrict__ Vb) {
  const int s = blockIdx.x;
  const int slot = blockIdx.y;  // 0-15: q heads, 16-23: k heads, 24-31: v heads
  const int d = threadIdx.x;    // 0..255

  int colBase;
  if (slot < 16)      colBase = slot * 256;
  else if (slot < 24) colBase = Q_SIZE + (slot - 16) * 256;
  else                colBase = Q_SIZE + KV_SIZE + (slot - 24) * 256;

  float x = qkv[(size_t)s * QKV_COLS + colBase + d];

  __shared__ float red[256];
  __shared__ float snorm[256];
  red[d] = x * x;
  __syncthreads();
  for (int off = 128; off > 0; off >>= 1) {
    if (d < off) red[d] += red[d + off];
    __syncthreads();
  }
  float var = red[0] * (1.0f / HEAD_DIM);
  float nx = x * rsqrtf(var + 1e-6f);
  if (slot < 16)      nx *= (1.0f + q_norm_w[d]);
  else if (slot < 24) nx *= (1.0f + k_norm_w[d]);
  snorm[d] = nx;
  __syncthreads();

  float y = nx;
  if (slot < 24 && d < ROT_DIM) {
    const int half = ROT_DIM / 2;  // 32
    float pos = (float)positions[s];
    int i = (d < half) ? d : d - half;
    float freq = pos * __powf(10000.0f, -((float)i) / (float)half);
    float sv, cv;
    __sincosf(freq, &sv, &cv);
    float x1 = snorm[(d < half) ? d : d - half];
    float x2 = snorm[(d < half) ? d + half : d];
    y = (d < half) ? (x1 * cv - x2 * sv) : (x2 * cv + x1 * sv);
  }
  unsigned short ob = f2bf(y);
  if (slot < 16)      Qb[((size_t)s * NUM_HEADS + slot) * HEAD_DIM + d] = ob;
  else if (slot < 24) Kb[((size_t)s * NUM_KV_HEADS + (slot - 16)) * HEAD_DIM + d] = ob;
  else                Vb[((size_t)s * NUM_KV_HEADS + (slot - 24)) * HEAD_DIM + d] = ob;
}

// ------------------------------------------------ Flash attention, one head/block
// K tile double-buffered via async engine; V transposed by VALU path (overlaps asyncs).
__global__ __launch_bounds__(256)
void attn_kernel(const unsigned short* __restrict__ Qb,
                 const unsigned short* __restrict__ Kb,
                 const unsigned short* __restrict__ Vb,
                 unsigned short* __restrict__ AO) {
  const int tid = threadIdx.x;
  const int lane = tid & 31, wave = tid >> 5;
  const int lsel = lane & 15, hi = lane >> 4;
  const int qb = blockIdx.x * 128;
  const int h = blockIdx.y;
  const int hk = h >> 1;  // 2 q heads per kv head

  __shared__ __align__(16) unsigned char Ks[2][32 * 528];  // K rows, padded, double-buffered
  __shared__ __align__(16) unsigned char VT[256 * 80];     // V transposed (d x kv)
  __shared__ __align__(16) unsigned char Ps[8 * 16 * 80];
  unsigned char* PW = Ps + wave * (16 * 80);

  const size_t kvrow = (size_t)NUM_KV_HEADS * HEAD_DIM;  // 2048 elems

  // this wave's 16 Q rows, all 256 dims, kept as 8 A-fragments in registers
  const int mrow = qb + wave * 16 + lsel;
  Frag qf[8];
#pragma unroll
  for (int dk = 0; dk < 8; dk++)
    qf[dk] = glb_frag_a(Qb + ((size_t)mrow * NUM_HEADS + h) * HEAD_DIM + dk * 32, hi);

  v8f O[16];
#pragma unroll
  for (int t = 0; t < 16; t++) O[t] = {};
  float m_run[8], l_run[8];
#pragma unroll
  for (int j = 0; j < 8; j++) { m_run[j] = -1e30f; l_run[j] = 0.0f; }

  auto stageK = [&](int buf, int n0) {
    const unsigned short* gk = Kb + (size_t)(n0 + wave) * kvrow + (size_t)hk * HEAD_DIM + lane * 8;
    unsigned char* dst = Ks[buf] + wave * 528 + lane * 16;
#pragma unroll
    for (int i = 0; i < 4; i++)
      async_cp16(gk + (size_t)(8 * i) * kvrow, dst + (8 * i) * 528);
  };

  const int kvEnd = qb + 128;  // causal: no kv beyond the last q row of this block
  stageK(0, 0);
  int p = 0;
  for (int n0 = 0; n0 < kvEnd; n0 += 32, p ^= 1) {
    if (n0 + 32 < kvEnd) stageK(p ^ 1, n0 + 32);  // prefetch next K tile

    // V tile transposed into LDS (b128 global reads, b16 scatter stores) — overlaps asyncs
#pragma unroll
    for (int i = 0; i < 4; i++) {
      int r = wave + 8 * i;
      const unsigned short* gv = Vb + (size_t)(n0 + r) * kvrow + (size_t)hk * HEAD_DIM + lane * 8;
      u32x4 vv = *(const u32x4*)gv;
#pragma unroll
      for (int j = 0; j < 4; j++) {
        unsigned int w = vv[j];
        *(unsigned short*)(VT + (lane * 8 + 2 * j) * 80 + r * 2)     = (unsigned short)(w & 0xFFFFu);
        *(unsigned short*)(VT + (lane * 8 + 2 * j + 1) * 80 + r * 2) = (unsigned short)(w >> 16);
      }
    }
    if (n0 + 32 < kvEnd) wait_async<4>();  // current K tile (first 4 ops) complete
    else                 wait_async<0>();
    __syncthreads();

    // S = Q * K^T  (two 16x16 tiles, K-depth 256)
    v8f s0 = {}, s1 = {};
#pragma unroll
    for (int dk = 0; dk < 8; dk++) {
      Frag b0 = lds_frag_b(Ks[p] + lsel * 528 + dk * 64, hi);
      Frag b1 = lds_frag_b(Ks[p] + (lsel + 16) * 528 + dk * 64, hi);
      s0 = __builtin_amdgcn_wmma_f32_16x16x32_bf16(false, qf[dk].bf, false, b0.bf, (short)0, s0, false, false);
      s1 = __builtin_amdgcn_wmma_f32_16x16x32_bf16(false, qf[dk].bf, false, b1.bf, (short)0, s1, false, false);
    }

    // causal mask + online softmax (rows spread over 16-lane groups)
    float p0[8], p1[8], scale[8];
#pragma unroll
    for (int j = 0; j < 8; j++) {
      int row = qb + wave * 16 + j + 8 * hi;
      int c0 = n0 + lsel, c1 = c0 + 16;
      float v0 = (c0 <= row) ? s0[j] : -1e30f;
      float v1 = (c1 <= row) ? s1[j] : -1e30f;
      float mx = fmaxf(v0, v1);
#pragma unroll
      for (int msk = 1; msk < 16; msk <<= 1)
        mx = fmaxf(mx, __shfl_xor(mx, msk, 32));
      float mnew = fmaxf(m_run[j], mx);
      scale[j] = __expf(m_run[j] - mnew);
      m_run[j] = mnew;
      p0[j] = __expf(v0 - mnew);
      p1[j] = __expf(v1 - mnew);
      float rs = p0[j] + p1[j];
#pragma unroll
      for (int msk = 1; msk < 16; msk <<= 1)
        rs += __shfl_xor(rs, msk, 32);
      l_run[j] = l_run[j] * scale[j] + rs;
    }
#pragma unroll
    for (int t = 0; t < 16; t++)
#pragma unroll
      for (int j = 0; j < 8; j++) O[t][j] = O[t][j] * scale[j];

    // P: C-layout -> A-layout via per-wave LDS bounce
#pragma unroll
    for (int j = 0; j < 8; j++) {
      int pr = j + 8 * hi;
      *(unsigned short*)(PW + pr * 80 + lsel * 2)        = f2bf(p0[j]);
      *(unsigned short*)(PW + pr * 80 + (lsel + 16) * 2) = f2bf(p1[j]);
    }
    asm volatile("s_wait_dscnt 0" ::: "memory");
    Frag pf = lds_frag_a(PW + lsel * 80, hi);

    // O += P * V   (16 output tiles across d)
#pragma unroll
    for (int dt = 0; dt < 16; dt++) {
      Frag vf = lds_frag_b(VT + (dt * 16 + lsel) * 80, hi);
      O[dt] = __builtin_amdgcn_wmma_f32_16x16x32_bf16(false, pf.bf, false, vf.bf, (short)0, O[dt], false, false);
    }
    __syncthreads();
  }

  // epilogue: divide by l, write bf16 into AO[s, h*256 + d]
#pragma unroll
  for (int dt = 0; dt < 16; dt++) {
    int col = h * HEAD_DIM + dt * 16 + lsel;
#pragma unroll
    for (int j = 0; j < 8; j++) {
      int row = qb + wave * 16 + j + 8 * hi;
      float val = O[dt][j] / l_run[j];
      AO[(size_t)row * Q_SIZE + col] = f2bf(val);
    }
  }
}

// ---------------------------------------------------------------- launcher
extern "C" void kernel_launch(void* const* d_in, const int* in_sizes, int n_in,
                              void* d_out, int out_size, void* d_ws, size_t ws_size,
                              hipStream_t stream) {
  (void)in_sizes; (void)n_in; (void)out_size; (void)ws_size;
  const int*   positions = (const int*)d_in[0];
  const float* hidden    = (const float*)d_in[1];
  const float* w_qkv     = (const float*)d_in[2];
  const float* q_norm_w  = (const float*)d_in[3];
  const float* k_norm_w  = (const float*)d_in[4];
  const float* w_o       = (const float*)d_in[5];
  float* out = (float*)d_out;

  unsigned char* ws = (unsigned char*)d_ws;
  size_t off = 0;
  auto alloc = [&](size_t bytes) {
    unsigned char* p = ws + off;
    off += (bytes + 255) & ~(size_t)255;
    return p;
  };
  unsigned short* hid_bf  = (unsigned short*)alloc((size_t)SEQ * HIDDEN * 2);
  unsigned short* wqkv_bf = (unsigned short*)alloc((size_t)QKV_COLS * HIDDEN * 2);
  unsigned short* wo_bf   = (unsigned short*)alloc((size_t)HIDDEN * Q_SIZE * 2);
  float*          qkv_f   = (float*)alloc((size_t)SEQ * QKV_COLS * 4);
  unsigned short* Qb      = (unsigned short*)alloc((size_t)SEQ * Q_SIZE * 2);
  unsigned short* Kb      = (unsigned short*)alloc((size_t)SEQ * KV_SIZE * 2);
  unsigned short* Vb      = (unsigned short*)alloc((size_t)SEQ * KV_SIZE * 2);
  unsigned short* AO      = (unsigned short*)alloc((size_t)SEQ * Q_SIZE * 2);

  long nh = (long)SEQ * HIDDEN;
  cvt_f32_bf16<<<(unsigned)(nh / 2048), 256, 0, stream>>>(hidden, hid_bf, nh);
  long nq = (long)QKV_COLS * HIDDEN;
  cvt_f32_bf16<<<(unsigned)(nq / 2048), 256, 0, stream>>>(w_qkv, wqkv_bf, nq);
  long no = (long)HIDDEN * Q_SIZE;
  cvt_f32_bf16<<<(unsigned)(no / 2048), 256, 0, stream>>>(w_o, wo_bf, no);

  // qkv = hidden @ w_qkv^T
  gemm_bf16_nt<<<dim3(QKV_COLS / 128, SEQ / 128), 256, 0, stream>>>(hid_bf, wqkv_bf, qkv_f, QKV_COLS, HIDDEN);
  // per-head rmsnorm + rope + bf16 pack
  norm_rope_kernel<<<dim3(SEQ, 32), 256, 0, stream>>>(qkv_f, q_norm_w, k_norm_w, positions, Qb, Kb, Vb);
  // flash attention
  attn_kernel<<<dim3(SEQ / 128, NUM_HEADS), 256, 0, stream>>>(Qb, Kb, Vb, AO);
  // out = AO @ w_o^T
  gemm_bf16_nt<<<dim3(HIDDEN / 128, SEQ / 128), 256, 0, stream>>>(AO, wo_bf, out, HIDDEN, Q_SIZE);
}
